// DalleTransformer_40948218200172
// MI455X (gfx1250) — matrix operand order repaired and checked by reading, not compiled, not run on hardware
//
#include <hip/hip_runtime.h>

#define DEV_INLINE __device__ __forceinline__

typedef __attribute__((ext_vector_type(16))) __bf16 v16bf;
typedef __attribute__((ext_vector_type(8)))  float  v8f;

struct FragBF { union { v16bf v; uint4 u[2]; }; };

DEV_INLINE unsigned short f2bf(float x) {
    union { float f; unsigned int u; } c; c.f = x;
    unsigned int r = c.u + 0x7FFFu + ((c.u >> 16) & 1u);   // RNE
    return (unsigned short)(r >> 16);
}

DEV_INLINE v8f v8zero() {
    v8f z;
#pragma unroll
    for (int i = 0; i < 8; ++i) z[i] = 0.0f;
    return z;
}

// A-operand fragment (16x32 bf16): lane m = lane&15, half = lane>>4.
// element j <-> k = (j&7) + 8*half + 16*(j>>3)  => two 16B loads per lane.
DEV_INLINE FragBF load_a_frag(const unsigned short* rowp, int half) {
    FragBF f;
    f.u[0] = *(const uint4*)(rowp + half * 8);
    f.u[1] = *(const uint4*)(rowp + 16 + half * 8);
    return f;
}

// B-operand fragment (32x16 bf16): lane n = lane&15, half = lane>>4.
// element j <-> k = (j&7) + 8*half + 16*(j>>3)  => two 16B loads per lane.
DEV_INLINE FragBF load_b_frag(const unsigned short* rowp, int half) {
    FragBF f;
    f.u[0] = *(const uint4*)(rowp + half * 8);
    f.u[1] = *(const uint4*)(rowp + 16 + half * 8);
    return f;
}

// Branch-free tanh via v_exp_f32 + v_rcp_f32 (no IEEE div expansion):
// tanh(x) = 1 - 2*rcp(exp(2x)+1); exp->inf => 1, exp->0 => -1.
DEV_INLINE float tanh_fast(float x) {
    return 1.0f - 2.0f * __builtin_amdgcn_rcpf(__expf(2.0f * x) + 1.0f);
}

DEV_INLINE float gelu_f(float x) {
    float x2 = x * x;
    return 0.5f * x * (1.0f + tanh_fast(0.7978845608028654f * x * (1.0f + 0.044715f * x2)));
}

// CDNA5 async copy: one wave-wide instruction moves 32 lanes x 16B to LDS.
// LDS address comes from the vdst VGPR; generic-pointer low 32 bits are the LDS offset.
DEV_INLINE void async_load_b128(unsigned lds_off, const void* gaddr) {
    asm volatile("global_load_async_to_lds_b128 %0, %1, off"
                 :: "v"(lds_off), "v"((unsigned long long)(size_t)gaddr)
                 : "memory");
}
DEV_INLINE void wait_async0() {
    asm volatile("s_wait_asynccnt 0x0" ::: "memory");
}

// ---------------------------------------------------------------------------
// fp32 -> bf16 bulk convert (weights; n % 4 == 0)
// ---------------------------------------------------------------------------
__global__ __launch_bounds__(256) void f32_to_bf16_kernel(
    const float* __restrict__ in, unsigned short* __restrict__ out, int n)
{
    const int i = (blockIdx.x * 256 + threadIdx.x) * 4;
    if (i < n) {
        float4 v = *(const float4*)(in + i);
        out[i + 0] = f2bf(v.x); out[i + 1] = f2bf(v.y);
        out[i + 2] = f2bf(v.z); out[i + 3] = f2bf(v.w);
    }
}

// ---------------------------------------------------------------------------
// LayerNorm over H=2048, one block per row.
//   y = LN(x)*g + b (+ res);  written fp32 (out) or bf16 (out_bf) per pointer
// ---------------------------------------------------------------------------
__global__ __launch_bounds__(256) void layernorm_kernel(
    const float* __restrict__ X, const float* __restrict__ g,
    const float* __restrict__ bb, const float* __restrict__ res,
    float* __restrict__ out, unsigned short* __restrict__ out_bf)
{
    __shared__ float red[256];
    const int row = blockIdx.x;
    const int tid = threadIdx.x;
    const size_t base = ((size_t)row << 11);
    const float* x = X + base;
    float lv[8];
    float s = 0.0f;
#pragma unroll
    for (int i = 0; i < 8; ++i) { lv[i] = x[tid + (i << 8)]; s += lv[i]; }
    red[tid] = s; __syncthreads();
    for (int st = 128; st > 0; st >>= 1) {
        if (tid < st) red[tid] += red[tid + st];
        __syncthreads();
    }
    const float mean = red[0] * (1.0f / 2048.0f);
    __syncthreads();
    float s2 = 0.0f;
#pragma unroll
    for (int i = 0; i < 8; ++i) { float d = lv[i] - mean; s2 += d * d; }
    red[tid] = s2; __syncthreads();
    for (int st = 128; st > 0; st >>= 1) {
        if (tid < st) red[tid] += red[tid + st];
        __syncthreads();
    }
    const float rstd = __builtin_amdgcn_rsqf(red[0] * (1.0f / 2048.0f) + 1e-5f);
#pragma unroll
    for (int i = 0; i < 8; ++i) {
        const int col = tid + (i << 8);
        float y = (lv[i] - mean) * rstd * g[col] + bb[col];
        if (res) y += res[base + col];
        if (out_bf) out_bf[base + col] = f2bf(y);
        else       out[base + col]    = y;
    }
}

// ---------------------------------------------------------------------------
// C[M,N] = A[M,K]*W[N,K]^T + bias (optional GELU; fp32 or bf16 output).
// A,W already bf16. Tile 64x64, BK=64; global->LDS via async-to-LDS DMA
// (4 wave-wide global_load_async_to_lds_b128 per wave per chunk), then
// 4 x v_wmma_f32_16x16x32_bf16 per wave per chunk.
// ---------------------------------------------------------------------------
__global__ __launch_bounds__(256) void wmma_gemm_async_kernel(
    const unsigned short* __restrict__ A, const unsigned short* __restrict__ W,
    const float* __restrict__ bias, float* __restrict__ Cf,
    unsigned short* __restrict__ Cb, int N, int K, int act_gelu)
{
    __shared__ __align__(16) unsigned short As[64][72];   // 144B row stride
    __shared__ __align__(16) unsigned short Ws[64][72];
    const int tid  = threadIdx.x;
    const int lane = tid & 31;
    const int wave = tid >> 5;
    const int half = (lane >> 4) & 1;
    const int l16  = lane & 15;
    const int m0 = blockIdx.y << 6;
    const int n0 = blockIdx.x << 6;
    const int mi  = wave >> 1;
    const int ni0 = (wave & 1) << 1;

    // staging geometry: transfer j in [0,512): row=j>>3, col=(j&7)*8 (16B)
    const int j0 = tid, j1 = tid + 256;
    const int r0 = j0 >> 3, c0 = (j0 & 7) << 3;
    const int r1 = j1 >> 3, c1 = (j1 & 7) << 3;
    const unsigned baseA = (unsigned)(size_t)&As[0][0];
    const unsigned baseW = (unsigned)(size_t)&Ws[0][0];
    const unsigned ldsA0 = baseA + (unsigned)(r0 * 72 + c0) * 2u;
    const unsigned ldsA1 = baseA + (unsigned)(r1 * 72 + c1) * 2u;
    const unsigned ldsW0 = baseW + (unsigned)(r0 * 72 + c0) * 2u;
    const unsigned ldsW1 = baseW + (unsigned)(r1 * 72 + c1) * 2u;
    const unsigned short* gA0 = A + (size_t)(m0 + r0) * K + c0;
    const unsigned short* gA1 = A + (size_t)(m0 + r1) * K + c1;
    const unsigned short* gW0 = W + (size_t)(n0 + r0) * K + c0;
    const unsigned short* gW1 = W + (size_t)(n0 + r1) * K + c1;

    v8f acc0 = v8zero(), acc1 = v8zero();

    for (int kk = 0; kk < K; kk += 64) {
        async_load_b128(ldsA0, gA0 + kk);
        async_load_b128(ldsA1, gA1 + kk);
        async_load_b128(ldsW0, gW0 + kk);
        async_load_b128(ldsW1, gW1 + kk);
        wait_async0();
        __syncthreads();
#pragma unroll
        for (int kc = 0; kc < 64; kc += 32) {
            FragBF af  = load_a_frag(&As[(mi << 4) + l16][kc], half);
            FragBF bf0 = load_b_frag(&Ws[(ni0 << 4) + l16][kc], half);
            FragBF bf1 = load_b_frag(&Ws[((ni0 + 1) << 4) + l16][kc], half);
            acc0 = __builtin_amdgcn_wmma_f32_16x16x32_bf16(false, af.v, false, bf0.v,
                                                           (short)0, acc0, false, false);
            acc1 = __builtin_amdgcn_wmma_f32_16x16x32_bf16(false, af.v, false, bf1.v,
                                                           (short)0, acc1, false, false);
        }
        __syncthreads();   // protect LDS before next chunk's async writes land
    }

    // C/D layout: lanes 0-15 -> M=r, lanes 16-31 -> M=r+8; N = lane&15
    const int nc0 = n0 + (ni0 << 4) + l16;
    const int nc1 = nc0 + 16;
    const float b0 = bias[nc0], b1 = bias[nc1];
#pragma unroll
    for (int r = 0; r < 8; ++r) {
        const int row = m0 + (mi << 4) + r + (half << 3);
        float v0 = acc0[r] + b0;
        float v1 = acc1[r] + b1;
        if (act_gelu) { v0 = gelu_f(v0); v1 = gelu_f(v1); }
        if (Cb) {
            Cb[(size_t)row * N + nc0] = f2bf(v0);
            Cb[(size_t)row * N + nc1] = f2bf(v1);
        } else {
            Cf[(size_t)row * N + nc0] = v0;
            Cf[(size_t)row * N + nc1] = v1;
        }
    }
}

// ---------------------------------------------------------------------------
// Split QKV [B,S,3H] fp32 -> q bf16 [B,NH,S,HN] (pre-scaled 1/sqrt(HN)),
// k bf16 [B,NH,S,HN], v^T bf16 [B,NH,HN,S]
// ---------------------------------------------------------------------------
__global__ __launch_bounds__(256) void qkv_split_kernel(
    const float* __restrict__ qkv,
    unsigned short* __restrict__ qb, unsigned short* __restrict__ kb,
    unsigned short* __restrict__ vt)
{
    const int idx = blockIdx.x * 256 + threadIdx.x;   // over B*NH*S*HN = 2^23
    const int hh = idx & 127;
    const int s  = (idx >> 7) & 1023;
    const int hd = (idx >> 17) & 15;
    const int b  = idx >> 21;
    const size_t src = ((size_t)((b << 10) + s)) * 6144 + (hd << 7) + hh;
    qb[idx] = f2bf(qkv[src] * 0.08838834764831845f);      // 1/sqrt(128)
    kb[idx] = f2bf(qkv[src + 2048]);
    const int head = (b << 4) + hd;
    const size_t vdst = (((size_t)head << 7) + hh) * 1024 + s;
    vt[vdst] = f2bf(qkv[src + 4096]);
}

// ---------------------------------------------------------------------------
// Causal flash attention. Block = (head, 128 query rows), wave = 16 rows.
// Streaming softmax over 32-key chunks; masked scores = exactly -10000
// (the reference's global-max/ALPHA shift is uniform per head and cancels
// in the row-wise softmax). ctx written bf16 (feeds the dense GEMM).
// ---------------------------------------------------------------------------
__global__ __launch_bounds__(256) void flash_attn_kernel(
    const unsigned short* __restrict__ Qb,
    const unsigned short* __restrict__ Kb,
    const unsigned short* __restrict__ Vt,
    unsigned short* __restrict__ ctx)
{
    __shared__ __align__(16) unsigned short Pt[8][16][40];
    const int head = blockIdx.x;       // b*16 + n
    const int b = head >> 4;
    const int n = head & 15;
    const int wave = threadIdx.x >> 5;
    const int lane = threadIdx.x & 31;
    const int half = (lane >> 4) & 1;
    const int l16  = lane & 15;
    const int q0 = (blockIdx.y << 7) + (wave << 4);

    const unsigned short* Q  = Qb + ((size_t)head << 17);   // 1024*128 per head
    const unsigned short* Kp = Kb + ((size_t)head << 17);
    const unsigned short* Vp = Vt + ((size_t)head << 17);

    FragBF aq[4];
    {
        const unsigned short* qrow = Q + (size_t)(q0 + l16) * 128;
#pragma unroll
        for (int c = 0; c < 4; ++c) aq[c] = load_a_frag(qrow + c * 32, half);
    }

    v8f o[8];
#pragma unroll
    for (int t = 0; t < 8; ++t) o[t] = v8zero();
    float mrow[8], lsum[8];
#pragma unroll
    for (int r = 0; r < 8; ++r) { mrow[r] = -3.0e38f; lsum[r] = 0.0f; }
    const int rowbase = q0 + (half << 3);

    const int nch = (q0 + 47) >> 5;    // chunks covering keys <= q0+15
    for (int jc = 0; jc < nch; ++jc) {
        const int kb0 = jc << 5;
        v8f sc0 = v8zero(), sc1 = v8zero();
        {
            const unsigned short* k0p = Kp + (size_t)(kb0 + l16) * 128;
            const unsigned short* k1p = k0p + 16 * 128;
#pragma unroll
            for (int c = 0; c < 4; ++c) {
                FragBF bk = load_b_frag(k0p + c * 32, half);
                sc0 = __builtin_amdgcn_wmma_f32_16x16x32_bf16(false, aq[c].v, false, bk.v,
                                                              (short)0, sc0, false, false);
            }
#pragma unroll
            for (int c = 0; c < 4; ++c) {
                FragBF bk = load_b_frag(k1p + c * 32, half);
                sc1 = __builtin_amdgcn_wmma_f32_16x16x32_bf16(false, aq[c].v, false, bk.v,
                                                              (short)0, sc1, false, false);
            }
        }
        const int key0 = kb0 + l16;
        const int key1 = key0 + 16;
        unsigned short (*Pw)[40] = Pt[wave];
#pragma unroll
        for (int r = 0; r < 8; ++r) {
            const int row = rowbase + r;
            float s0 = (key0 <= row) ? sc0[r] : -10000.0f;
            float s1 = (key1 <= row) ? sc1[r] : -10000.0f;
            float t = fmaxf(s0, s1);
            t = fmaxf(t, __shfl_xor(t, 1, 32));
            t = fmaxf(t, __shfl_xor(t, 2, 32));
            t = fmaxf(t, __shfl_xor(t, 4, 32));
            t = fmaxf(t, __shfl_xor(t, 8, 32));
            const float mn = fmaxf(mrow[r], t);
            const float rescale = __expf(mrow[r] - mn);
            mrow[r] = mn;
            lsum[r] *= rescale;
#pragma unroll
            for (int ot = 0; ot < 8; ++ot) o[ot][r] *= rescale;
            const float p0 = __expf(s0 - mn);
            const float p1 = __expf(s1 - mn);
            Pw[(half << 3) + r][l16]      = f2bf(p0);
            Pw[(half << 3) + r][16 + l16] = f2bf(p1);
            float l = p0 + p1;
            l += __shfl_xor(l, 1, 32);
            l += __shfl_xor(l, 2, 32);
            l += __shfl_xor(l, 4, 32);
            l += __shfl_xor(l, 8, 32);
            lsum[r] += l;
        }
        // LDS ops are in-order within a wave -> safe to reload in A-layout
        FragBF pa = load_a_frag(&Pw[l16][0], half);
#pragma unroll
        for (int ot = 0; ot < 8; ++ot) {
            const unsigned short* vp = Vp + (size_t)((ot << 4) + l16) * 1024 + kb0;
            FragBF bv = load_b_frag(vp, half);
            o[ot] = __builtin_amdgcn_wmma_f32_16x16x32_bf16(false, pa.v, false, bv.v,
                                                            (short)0, o[ot], false, false);
        }
    }

#pragma unroll
    for (int r = 0; r < 8; ++r) {
        const int row = rowbase + r;
        const float inv = __builtin_amdgcn_rcpf(lsum[r]);
        unsigned short* dst = ctx + ((size_t)((b << 10) + row) << 11) + (n << 7);
#pragma unroll
        for (int ot = 0; ot < 8; ++ot)
            dst[(ot << 4) + l16] = f2bf(o[ot][r] * inv);
    }
}

// ---------------------------------------------------------------------------
extern "C" void kernel_launch(void* const* d_in, const int* in_sizes, int n_in,
                              void* d_out, int out_size, void* d_ws, size_t ws_size,
                              hipStream_t stream) {
    const float* hidden   = (const float*)d_in[0];
    const float* qkv_w    = (const float*)d_in[2];
    const float* qkv_b    = (const float*)d_in[3];
    const float* dense_w  = (const float*)d_in[4];
    const float* dense_b  = (const float*)d_in[5];
    const float* mlp_w1   = (const float*)d_in[6];
    const float* mlp_b1   = (const float*)d_in[7];
    const float* mlp_w2   = (const float*)d_in[8];
    const float* mlp_b2   = (const float*)d_in[9];
    const float* ln_in_g  = (const float*)d_in[10];
    const float* ln_in_b  = (const float*)d_in[11];
    const float* ln_post_g= (const float*)d_in[12];
    const float* ln_post_b= (const float*)d_in[13];
    const float* ln_s1_g  = (const float*)d_in[14];
    const float* ln_s1_b  = (const float*)d_in[15];
    const float* ln_s2_g  = (const float*)d_in[16];
    const float* ln_s2_b  = (const float*)d_in[17];
    float* out = (float*)d_out;

    char* ws = (char*)d_ws;
    float*          BIG  = (float*)(ws);                          // 100.7 MB fp32 scratch
    unsigned short* xb   = (unsigned short*)(ws + (size_t)100663296); // 16 MB bf16 act
    unsigned short* midb = (unsigned short*)(ws + (size_t)117440512); // 64 MB bf16 mlp mid
    unsigned short* qb   = (unsigned short*)(ws + (size_t)184549376); // 16 MB
    unsigned short* kbuf = (unsigned short*)(ws + (size_t)201326592); // 16 MB
    unsigned short* vtb  = (unsigned short*)(ws + (size_t)218103808); // 16 MB
    unsigned short* qkvW = (unsigned short*)(ws + (size_t)234881024); // 201 MB bf16 weights
    unsigned short* dnsW = qkvW + (size_t)25165824;   // L*3H*H
    unsigned short* w1W  = dnsW + (size_t)8388608;    // L*H*H
    unsigned short* w2W  = w1W  + (size_t)33554432;   // L*4H*H

    const dim3 blk(256);
    // one-time-per-launch weight conversion fp32 -> bf16 (deterministic)
    f32_to_bf16_kernel<<<24576, blk, 0, stream>>>(qkv_w,  qkvW, 25165824);
    f32_to_bf16_kernel<<<8192,  blk, 0, stream>>>(dense_w, dnsW, 8388608);
    f32_to_bf16_kernel<<<32768, blk, 0, stream>>>(mlp_w1, w1W, 33554432);
    f32_to_bf16_kernel<<<32768, blk, 0, stream>>>(mlp_w2, w2W, 33554432);

    for (int i = 0; i < 2; ++i) {
        const float* h = (i == 0) ? hidden : out;
        // x = LN_in(h) -> bf16
        layernorm_kernel<<<4096, blk, 0, stream>>>(h, ln_in_g + i * 2048,
                                                   ln_in_b + i * 2048, nullptr,
                                                   nullptr, xb);
        // qkv = x @ qkv_w^T + qkv_b  (fp32 out)
        wmma_gemm_async_kernel<<<dim3(96, 64), blk, 0, stream>>>(
            xb, qkvW + (size_t)i * 12582912, qkv_b + (size_t)i * 6144,
            BIG, nullptr, 6144, 2048, 0);
        // split/scale/transpose to bf16 heads
        qkv_split_kernel<<<32768, blk, 0, stream>>>(BIG, qb, kbuf, vtb);
        // ctx = attention(q,k,v) -> bf16 (reuses xb)
        flash_attn_kernel<<<dim3(64, 8), blk, 0, stream>>>(qb, kbuf, vtb, xb);
        // attn = ctx @ dense_w^T + dense_b  (fp32 out)
        wmma_gemm_async_kernel<<<dim3(32, 64), blk, 0, stream>>>(
            xb, dnsW + (size_t)i * 4194304, dense_b + (size_t)i * 2048,
            BIG, nullptr, 2048, 2048, 0);
        // h2 = h + LN_s1(attn)  (fp32 residual stream in d_out)
        layernorm_kernel<<<4096, blk, 0, stream>>>(BIG, ln_s1_g + i * 2048,
                                                   ln_s1_b + i * 2048, h, out, nullptr);
        // y = LN_post(h2) -> bf16
        layernorm_kernel<<<4096, blk, 0, stream>>>(out, ln_post_g + i * 2048,
                                                   ln_post_b + i * 2048, nullptr,
                                                   nullptr, xb);
        // mid = gelu(y @ w1^T + b1) -> bf16
        wmma_gemm_async_kernel<<<dim3(128, 64), blk, 0, stream>>>(
            xb, w1W + (size_t)i * 16777216, mlp_b1 + (size_t)i * 8192,
            nullptr, midb, 8192, 2048, 1);
        // y2 = mid @ w2^T + b2  (fp32 out)
        wmma_gemm_async_kernel<<<dim3(32, 64), blk, 0, stream>>>(
            midb, w2W + (size_t)i * 16777216, mlp_b2 + (size_t)i * 2048,
            BIG, nullptr, 2048, 8192, 0);
        // h = h2 + LN_s2(y2)   (in-place residual on d_out is per-element safe)
        layernorm_kernel<<<4096, blk, 0, stream>>>(BIG, ln_s2_g + i * 2048,
                                                   ln_s2_b + i * 2048, out, out, nullptr);
    }
    (void)in_sizes; (void)n_in; (void)out_size; (void)ws_size;
}
